// KGEncoder_83657372991995
// MI455X (gfx1250) — matrix operand order repaired and checked by reading, not compiled, not run on hardware
//
#include <hip/hip_runtime.h>
#include <hip/hip_bf16.h>
#include <math.h>
#include <stdint.h>

// ---------------- problem constants (match reference) ----------------
#define N_NODES 20000
#define E_EDGES 120000
#define G_GR    64
#define D_DIM   512
#define DE_DIM  256
#define H_HEADS 8
#define Q_NQ    3
#define K_CB    512
#define FF_DIM  2048
#define L_LAYERS 3
#define C_HD    64          // D / H
#define D4      (4 * D_DIM) // 2048

#define KCHUNK   512        // K-extent staged in LDS per pass
#define BPAD     8          // bf16 elements of row padding (16 B)
#define BSTRIDE  (KCHUNK + BPAD)

typedef __attribute__((ext_vector_type(16))) __bf16 v16bf;
typedef __attribute__((ext_vector_type(8)))  __bf16 v8bf;
typedef __attribute__((ext_vector_type(8)))  float  v8f;

// =====================================================================
// fp32 -> bf16 bulk convert (8 elements / thread, vectorized)
// =====================================================================
__global__ __launch_bounds__(256) void f32_to_bf16_kernel(
    const float* __restrict__ src, __bf16* __restrict__ dst, int n8)
{
    int i = blockIdx.x * blockDim.x + threadIdx.x;
    if (i >= n8) return;
    const float* s = src + (size_t)i * 8;
    v8bf o;
#pragma unroll
    for (int j = 0; j < 8; j++) o[j] = (__bf16)s[j];
    *(v8bf*)(dst + (size_t)i * 8) = o;
}

// =====================================================================
// W[K,N] fp32 -> Wt[N,K] bf16 (tiled transpose through LDS)
// grid: (K/32, N/32), 256 threads = 32x8
// =====================================================================
__global__ __launch_bounds__(256) void transpose_bf16_kernel(
    const float* __restrict__ W, __bf16* __restrict__ Wt, int K, int N)
{
    __shared__ float tile[32][33];
    int kb = blockIdx.x * 32, nb = blockIdx.y * 32;
    int tx = threadIdx.x & 31, ty = threadIdx.x >> 5;
#pragma unroll
    for (int i = ty; i < 32; i += 8)
        tile[i][tx] = W[(size_t)(kb + i) * N + nb + tx];
    __syncthreads();
#pragma unroll
    for (int i = ty; i < 32; i += 8)
        Wt[(size_t)(nb + i) * K + kb + tx] = (__bf16)tile[tx][i];
}

// =====================================================================
// async global -> LDS helpers (CDNA5 async path, ASYNCcnt-tracked)
// =====================================================================
__device__ __forceinline__ void async_load_b128(uint32_t lds_off, uint64_t gaddr)
{
    asm volatile("global_load_async_to_lds_b128 %0, %1, off"
                 :: "v"(lds_off), "v"(gaddr) : "memory");
}
__device__ __forceinline__ void wait_async0()
{
    asm volatile("s_wait_asynccnt 0x0" ::: "memory");
}

// =====================================================================
// WMMA bf16 GEMM:  C[M,N] = A[M,K] @ Wt[N,K]^T + bias[N] (opt ReLU)
// A bf16 row-major, Wt bf16 [N,K] (pre-transposed weights).
// Block: 256 thr = 8 waves; block tile 128(M) x 32(N); each wave owns a
// 16x32 strip (two accumulators -> 2 back-to-back WMMAs share one A
// fragment, halving A traffic). B slab (32 cols x K-chunk) staged in
// LDS via async DMA; all fragment reads are aligned b128 loads.
// =====================================================================
__global__ __launch_bounds__(256) void gemm_bf16_wmma(
    const __bf16* __restrict__ A,
    const __bf16* __restrict__ Bt,
    const float* __restrict__ bias,
    float* __restrict__ C,
    int M, int K, int N, int relu)
{
    const int ntiles = N >> 5;              // 32-column tiles
    const int mtile  = blockIdx.x / ntiles;
    const int ntile  = blockIdx.x % ntiles;
    const int wave   = threadIdx.x >> 5;
    const int lane   = threadIdx.x & 31;
    const int row0   = mtile * 128 + wave * 16;
    const int col0   = ntile << 5;

    __shared__ __bf16 Bs[32 * BSTRIDE];     // 32 cols x K-chunk

    const int m  = lane & 15;
    const int hi = lane >> 4;      // 0 or 1
    int arow = row0 + m;
    if (arow >= M) arow = M - 1;   // clamp: garbage rows never stored
    const __bf16* Arow = A + (size_t)arow * K;

    v8f acc0 = {}, acc1 = {};

    for (int kc = 0; kc < K; kc += KCHUNK) {
        const int kcl = (K - kc < KCHUNK) ? (K - kc) : KCHUNK;

        // ---- async stage: Bs[r][0..kcl-1] = Bt[col0+r][kc..kc+kcl-1]
        {
            const int opsPerRow = kcl >> 3;          // 8 bf16 (16 B) per op
            const int nops = opsPerRow * 32;
            for (int t = threadIdx.x; t < nops; t += 256) {
                int r  = t / opsPerRow;
                int cb = (t - r * opsPerRow) << 3;
                uint32_t ldsoff = (uint32_t)(uintptr_t)&Bs[r * BSTRIDE + cb];
                uint64_t g = (uint64_t)(uintptr_t)(Bt + (size_t)(col0 + r) * K + kc + cb);
                async_load_b128(ldsoff, g);
            }
            wait_async0();
        }
        __syncthreads();

        for (int k0 = 0; k0 < kcl; k0 += 32) {
            if (k0 + 64 < kcl) __builtin_prefetch(Arow + kc + k0 + 64, 0, 1);

            // A fragment (contiguous K, matches B):
            // lanes 0-15 : elems 0..15 = K+0..15
            // lanes 16-31: elems 0..15 = K+16..31
            const int kbase = kc + k0 + hi * 16;
            v8bf a_lo = *(const v8bf*)(Arow + kbase);
            v8bf a_hi = *(const v8bf*)(Arow + kbase + 8);
            v16bf afrag = __builtin_shufflevector(a_lo, a_hi,
                0, 1, 2, 3, 4, 5, 6, 7, 8, 9, 10, 11, 12, 13, 14, 15);

            // B fragments: lane's columns = col0+(lane&15) and +16;
            // lanes 0-15 take K=k0..k0+15, lanes 16-31 take K=k0+16..k0+31
            const __bf16* brow0 = &Bs[(lane & 15) * BSTRIDE + k0 + hi * 16];
            const __bf16* brow1 = brow0 + 16 * BSTRIDE;
            v8bf b0lo = *(const v8bf*)(brow0);
            v8bf b0hi = *(const v8bf*)(brow0 + 8);
            v8bf b1lo = *(const v8bf*)(brow1);
            v8bf b1hi = *(const v8bf*)(brow1 + 8);
            v16bf bfrag0 = __builtin_shufflevector(b0lo, b0hi,
                0, 1, 2, 3, 4, 5, 6, 7, 8, 9, 10, 11, 12, 13, 14, 15);
            v16bf bfrag1 = __builtin_shufflevector(b1lo, b1hi,
                0, 1, 2, 3, 4, 5, 6, 7, 8, 9, 10, 11, 12, 13, 14, 15);

            acc0 = __builtin_amdgcn_wmma_f32_16x16x32_bf16(
                    false, afrag, false, bfrag0, (short)0, acc0, false, false);
            acc1 = __builtin_amdgcn_wmma_f32_16x16x32_bf16(
                    false, afrag, false, bfrag1, (short)0, acc1, false, false);
        }
        __syncthreads();   // before next chunk overwrites Bs
    }

    // C layout: VGPR r, lanes 0-15 -> M=r ; lanes 16-31 -> M=8+r
    const int crow0 = row0 + hi * 8;
    const int ccol0 = col0 + (lane & 15);
    const int ccol1 = ccol0 + 16;
    const float bv0 = bias[ccol0];
    const float bv1 = bias[ccol1];
#pragma unroll
    for (int r = 0; r < 8; r++) {
        int crow = crow0 + r;
        if (crow < M) {
            float v0 = acc0[r] + bv0;
            float v1 = acc1[r] + bv1;
            if (relu) { v0 = fmaxf(v0, 0.f); v1 = fmaxf(v1, 0.f); }
            float* crow_p = C + (size_t)crow * N;
            crow_p[ccol0] = v0;
            crow_p[ccol1] = v1;
        }
    }
}

// =====================================================================
// small helpers
// =====================================================================
__global__ void fill_kernel(float* p, float v, int n)
{
    int i = blockIdx.x * blockDim.x + threadIdx.x;
    if (i < n) p[i] = v;
}

__global__ void count_kernel(const int* __restrict__ batch, float* cnt)
{
    int i = blockIdx.x * blockDim.x + threadIdx.x;
    if (i < N_NODES) atomicAdd(&cnt[batch[i]], 1.0f);
}

__device__ __forceinline__ void atomicMaxF(float* addr, float val)
{
    if (val >= 0.f) atomicMax((int*)addr, __float_as_int(val));
    else            atomicMin((unsigned int*)addr, __float_as_uint(val));
}

// =====================================================================
// edge attention
// =====================================================================
__global__ void edge_alpha_kernel(const float* __restrict__ q,
                                  const float* __restrict__ k,
                                  const float* __restrict__ ep,
                                  const int* __restrict__ src,
                                  const int* __restrict__ dst,
                                  float* __restrict__ alpha,
                                  float* __restrict__ mmax)
{
    int tid = blockIdx.x * blockDim.x + threadIdx.x;
    if (tid >= E_EDGES * H_HEADS) return;
    int e = tid / H_HEADS, h = tid % H_HEADS;
    int s = src[e], d = dst[e];
    const float* qp  = q  + (size_t)d * D_DIM + h * C_HD;
    const float* kp  = k  + (size_t)s * D_DIM + h * C_HD;
    const float* epp = ep + (size_t)e * D_DIM + h * C_HD;
    float acc = 0.f;
#pragma unroll 8
    for (int c = 0; c < C_HD; c++) acc += qp[c] * (kp[c] + epp[c]);
    acc *= 0.125f;  // 1/sqrt(64)
    alpha[tid] = acc;
    atomicMaxF(&mmax[(size_t)d * H_HEADS + h], acc);
}

__global__ void edge_exp_kernel(float* __restrict__ alpha,
                                const float* __restrict__ mmax,
                                float* __restrict__ ssum,
                                const int* __restrict__ dst)
{
    int tid = blockIdx.x * blockDim.x + threadIdx.x;
    if (tid >= E_EDGES * H_HEADS) return;
    int e = tid / H_HEADS, h = tid % H_HEADS;
    int d = dst[e];
    float a = __expf(alpha[tid] - mmax[(size_t)d * H_HEADS + h]);
    alpha[tid] = a;
    atomicAdd(&ssum[(size_t)d * H_HEADS + h], a);
}

__global__ void edge_agg_kernel(const float* __restrict__ alpha,
                                const float* __restrict__ ssum,
                                const float* __restrict__ v,
                                const float* __restrict__ ep,
                                const int* __restrict__ src,
                                const int* __restrict__ dst,
                                float* __restrict__ agg)
{
    int tid = blockIdx.x * blockDim.x + threadIdx.x;
    if (tid >= E_EDGES * H_HEADS) return;
    int e = tid / H_HEADS, h = tid % H_HEADS;
    int s = src[e], d = dst[e];
    float w = alpha[tid] / (ssum[(size_t)d * H_HEADS + h] + 1e-16f);
    const float* vp  = v  + (size_t)s * D_DIM + h * C_HD;
    const float* epp = ep + (size_t)e * D_DIM + h * C_HD;
    float* ap = agg + (size_t)d * D_DIM + h * C_HD;
#pragma unroll 4
    for (int c = 0; c < C_HD; c++)
        atomicAdd(&ap[c], w * (vp[c] + epp[c]));
}

// =====================================================================
// residual + GraphNorm
// =====================================================================
__global__ void node_h_kernel(const float* __restrict__ agg,
                              const float* __restrict__ skp,
                              const float* __restrict__ x_in,
                              float* __restrict__ hbuf,
                              float* __restrict__ gsum,
                              const int* __restrict__ batch)
{
    int tid = blockIdx.x * blockDim.x + threadIdx.x;
    if (tid >= N_NODES * D_DIM) return;
    int n = tid / D_DIM, d = tid % D_DIM;
    float xv = agg[tid] + skp[tid];     // skp already includes bskip
    float h  = xv + x_in[tid];
    hbuf[tid] = h;
    atomicAdd(&gsum[(size_t)batch[n] * D_DIM + d], h);
}

__global__ void gn_out_kernel(const float* __restrict__ hbuf,
                              const float* __restrict__ gsum,
                              const float* __restrict__ cnt,
                              const float* __restrict__ gn_m,
                              float* __restrict__ obuf,
                              float* __restrict__ gvar,
                              const int* __restrict__ batch)
{
    int tid = blockIdx.x * blockDim.x + threadIdx.x;
    if (tid >= N_NODES * D_DIM) return;
    int n = tid / D_DIM, d = tid % D_DIM;
    int g = batch[n];
    float cg   = fmaxf(cnt[g], 1.f);
    float mean = gsum[(size_t)g * D_DIM + d] / cg;
    float out  = hbuf[tid] - gn_m[d] * mean;
    obuf[tid] = out;
    atomicAdd(&gvar[(size_t)g * D_DIM + d], out * out);
}

__global__ void gn_fin_kernel(const float* __restrict__ obuf,
                              const float* __restrict__ gvar,
                              const float* __restrict__ cnt,
                              const float* __restrict__ gn_w,
                              const float* __restrict__ gn_b,
                              float* __restrict__ x_cur,
                              const int* __restrict__ batch)
{
    int tid = blockIdx.x * blockDim.x + threadIdx.x;
    if (tid >= N_NODES * D_DIM) return;
    int n = tid / D_DIM, d = tid % D_DIM;
    int g = batch[n];
    float cg  = fmaxf(cnt[g], 1.f);
    float var = gvar[(size_t)g * D_DIM + d] / cg;
    x_cur[tid] = obuf[tid] * rsqrtf(var + 1e-5f) * gn_w[d] + gn_b[d];
}

__global__ void pool_kernel(const float* __restrict__ x_cur,
                            float* __restrict__ pooled,
                            const int* __restrict__ batch)
{
    int tid = blockIdx.x * blockDim.x + threadIdx.x;
    if (tid >= N_NODES * D_DIM) return;
    int n = tid / D_DIM, d = tid % D_DIM;
    atomicAdd(&pooled[(size_t)batch[n] * D_DIM + d], x_cur[tid]);
}

// =====================================================================
// Residual VQ (one block per graph)
// =====================================================================
__global__ void rvq_kernel(const float* __restrict__ pooled,
                           const float* __restrict__ cnt,
                           const float* __restrict__ codebooks,
                           float* __restrict__ tokens,
                           float* __restrict__ idxout,
                           float* __restrict__ lossbuf)
{
    int g = blockIdx.x;
    int t = threadIdx.x;   // 256 threads
    __shared__ float res[D_DIM];
    __shared__ float codebest[D_DIM];
    __shared__ float red[256];
    __shared__ int   redi[256];

    float cg = fmaxf(cnt[g], 1.f);
    for (int d = t; d < D_DIM; d += 256)
        res[d] = pooled[(size_t)g * D_DIM + d] / cg;
    __syncthreads();

    for (int qi = 0; qi < Q_NQ; qi++) {
        float s = 0.f;
        for (int d = t; d < D_DIM; d += 256) s += res[d] * res[d];
        red[t] = s; __syncthreads();
        for (int o = 128; o > 0; o >>= 1) {
            if (t < o) red[t] += red[t + o];
            __syncthreads();
        }
        float rnorm = fmaxf(sqrtf(red[0]), 1e-12f);
        __syncthreads();

        float best = -3.4e38f; int besti = 0;
        for (int k = t; k < K_CB; k += 256) {
            const float* cb = codebooks + ((size_t)qi * K_CB + k) * D_DIM;
            float dot = 0.f, nrm = 0.f;
            for (int d = 0; d < D_DIM; d++) {
                float c = cb[d];
                dot += res[d] * c;
                nrm += c * c;
            }
            float sim = dot / (rnorm * fmaxf(sqrtf(nrm), 1e-12f));
            if (sim > best || (sim == best && k < besti)) { best = sim; besti = k; }
        }
        red[t] = best; redi[t] = besti; __syncthreads();
        for (int o = 128; o > 0; o >>= 1) {
            if (t < o) {
                if (red[t + o] > red[t] ||
                    (red[t + o] == red[t] && redi[t + o] < redi[t])) {
                    red[t] = red[t + o]; redi[t] = redi[t + o];
                }
            }
            __syncthreads();
        }
        int kbest = redi[0];
        __syncthreads();

        const float* cb = codebooks + ((size_t)qi * K_CB + kbest) * D_DIM;
        float s2 = 0.f;
        for (int d = t; d < D_DIM; d += 256) s2 += cb[d] * cb[d];
        red[t] = s2; __syncthreads();
        for (int o = 128; o > 0; o >>= 1) {
            if (t < o) red[t] += red[t + o];
            __syncthreads();
        }
        float cn = fmaxf(sqrtf(red[0]), 1e-12f);
        __syncthreads();

        float commit = 0.f;
        for (int d = t; d < D_DIM; d += 256) {
            float code = cb[d] / cn;
            codebest[d] = code;
            tokens[((size_t)g * Q_NQ + qi) * D_DIM + d] = code;
            float df = code - res[d];
            commit += df * df;
        }
        red[t] = commit; __syncthreads();
        for (int o = 128; o > 0; o >>= 1) {
            if (t < o) red[t] += red[t + o];
            __syncthreads();
        }
        if (t == 0) {
            atomicAdd(&lossbuf[qi], red[0]);
            idxout[g * Q_NQ + qi] = (float)kbest;
        }
        __syncthreads();
        for (int d = t; d < D_DIM; d += 256) res[d] -= codebest[d];
        __syncthreads();
    }

    float s3 = 0.f;
    for (int d = t; d < D_DIM; d += 256) s3 += res[d] * res[d];
    red[t] = s3; __syncthreads();
    for (int o = 128; o > 0; o >>= 1) {
        if (t < o) red[t] += red[t + o];
        __syncthreads();
    }
    if (t == 0) atomicAdd(&lossbuf[Q_NQ], red[0]);
}

// =====================================================================
// t = LN(proj + skip) per token row
// =====================================================================
__global__ void add_ln_kernel(const float* __restrict__ proj,
                              const float* __restrict__ skf,
                              const float* __restrict__ ln_g,
                              const float* __restrict__ ln_b,
                              float* __restrict__ outp)
{
    int row = blockIdx.x;
    int t   = threadIdx.x;
    __shared__ float red[256], red2[256];
    const float* pp = proj + (size_t)row * FF_DIM;
    const float* sp = skf  + (size_t)row * FF_DIM;
    float s = 0.f, s2 = 0.f;
    for (int f = t; f < FF_DIM; f += 256) {
        float v = pp[f] + sp[f];
        s += v; s2 += v * v;
    }
    red[t] = s; red2[t] = s2; __syncthreads();
    for (int o = 128; o > 0; o >>= 1) {
        if (t < o) { red[t] += red[t + o]; red2[t] += red2[t + o]; }
        __syncthreads();
    }
    float mu  = red[0] / FF_DIM;
    float var = red2[0] / FF_DIM - mu * mu;
    float inv = rsqrtf(var + 1e-5f);
    for (int f = t; f < FF_DIM; f += 256) {
        float v = pp[f] + sp[f];
        outp[(size_t)row * FF_DIM + f] = (v - mu) * inv * ln_g[f] + ln_b[f];
    }
}

// =====================================================================
// per-graph renorm over (Q,FF) with unbiased std, * 0.02 + kg_bias
// =====================================================================
__global__ void renorm_kernel(const float* __restrict__ tn,
                              const float* __restrict__ kg_bias,
                              float* __restrict__ outp)
{
    int g = blockIdx.x;
    int t = threadIdx.x;
    const int n = Q_NQ * FF_DIM;
    __shared__ float red[256], red2[256];
    const float* p = tn + (size_t)g * n;
    float s = 0.f, s2 = 0.f;
    for (int i = t; i < n; i += 256) { float v = p[i]; s += v; s2 += v * v; }
    red[t] = s; red2[t] = s2; __syncthreads();
    for (int o = 128; o > 0; o >>= 1) {
        if (t < o) { red[t] += red[t + o]; red2[t] += red2[t + o]; }
        __syncthreads();
    }
    float mu   = red[0] / n;
    float ss   = red2[0] - (float)n * mu * mu;
    float stdv = fmaxf(sqrtf(ss / (float)(n - 1)), 1e-6f);
    float inv  = 0.02f / stdv;
    for (int i = t; i < n; i += 256) {
        int f = i % FF_DIM;
        outp[(size_t)g * n + i] = (p[i] - mu) * inv + kg_bias[f];
    }
}

__global__ void finalize_kernel(const float* __restrict__ lossbuf,
                                float* __restrict__ outp)
{
    if (blockIdx.x == 0 && threadIdx.x == 0) {
        float denom  = (float)G_GR * (float)D_DIM;
        float commit = (lossbuf[0] + lossbuf[1] + lossbuf[2]) / ((float)Q_NQ * denom);
        float rl     = lossbuf[3] / denom;
        outp[0] = 0.25f * commit + rl;
    }
}

// =====================================================================
// host orchestration
// =====================================================================
extern "C" void kernel_launch(void* const* d_in, const int* in_sizes, int n_in,
                              void* d_out, int out_size, void* d_ws, size_t ws_size,
                              hipStream_t stream)
{
    (void)in_sizes; (void)n_in; (void)out_size; (void)ws_size;

    const float* x         = (const float*)d_in[0];
    const float* edge_attr = (const float*)d_in[1];
    const float* Wq   = (const float*)d_in[2];
    const float* bq   = (const float*)d_in[3];
    const float* Wk   = (const float*)d_in[4];
    const float* bk   = (const float*)d_in[5];
    const float* Wv   = (const float*)d_in[6];
    const float* bv   = (const float*)d_in[7];
    const float* We   = (const float*)d_in[8];
    const float* be   = (const float*)d_in[9];
    const float* Wsk  = (const float*)d_in[10];
    const float* bsk  = (const float*)d_in[11];
    const float* gn_w = (const float*)d_in[12];
    const float* gn_b = (const float*)d_in[13];
    const float* gn_m = (const float*)d_in[14];
    const float* codebooks = (const float*)d_in[15];
    const float* Wsf  = (const float*)d_in[16];
    const float* bsf  = (const float*)d_in[17];
    const float* W1   = (const float*)d_in[18];
    const float* b1   = (const float*)d_in[19];
    const float* W2   = (const float*)d_in[20];
    const float* b2   = (const float*)d_in[21];
    const float* ln_g = (const float*)d_in[22];
    const float* ln_b = (const float*)d_in[23];
    const float* kg_bias = (const float*)d_in[24];
    const int*   edge_index = (const int*)d_in[25];
    const int*   batch      = (const int*)d_in[26];
    const int*   src = edge_index;
    const int*   dst = edge_index + E_EDGES;

    float* outp = (float*)d_out;

    // ---- sizes ----
    const size_t ND  = (size_t)N_NODES * D_DIM;
    const size_t ED  = (size_t)E_EDGES * D_DIM;
    const size_t EDe = (size_t)E_EDGES * DE_DIM;
    const size_t EH  = (size_t)E_EDGES * H_HEADS;
    const size_t NH  = (size_t)N_NODES * H_HEADS;
    const size_t GD  = (size_t)G_GR * D_DIM;
    const size_t GQD = (size_t)G_GR * Q_NQ * D_DIM;
    const size_t GQF = (size_t)G_GR * Q_NQ * FF_DIM;
    const int    GQ  = G_GR * Q_NQ;   // 192

    // ---- workspace carve (in floats; bf16 buffers use half) ----
    float* ws = (float*)d_ws;
    size_t off = 0;
    auto carve = [&](size_t nfloats) { float* p = ws + off; off += nfloats; return p; };
    float* x_cur  = carve(ND);
    float* qb     = carve(ND);
    float* kb     = carve(ND);
    float* vb     = carve(ND);
    float* skipb  = carve(ND);
    float* aggb   = carve(ND);
    float* hbuf   = carve(ND);
    float* obuf   = carve(ND);
    float* eprojb = carve(ED);
    float* alphab = carve(EH);
    float* mb     = carve(NH);
    float* ssumb  = carve(NH);
    float* gsum   = carve(GD);
    float* gvar   = carve(GD);
    float* pooled = carve(GD);
    float* cntb   = carve(G_GR);
    float* tokens = carve(GQD);
    float* h1b    = carve((size_t)GQ * D4);
    float* projb  = carve(GQF);
    float* skfb   = carve(GQF);
    float* tnorm  = carve(GQF);
    float* lossb  = carve(8);
    __bf16* xbf  = (__bf16*)carve(ND / 2 + 16);           // x in bf16
    __bf16* ebf  = (__bf16*)carve(EDe / 2 + 16);          // edge_attr bf16
    __bf16* tkbf = (__bf16*)carve(GQD / 2 + 16);          // tokens bf16
    __bf16* h1bf = (__bf16*)carve((size_t)GQ * D4 / 2 + 16);
    __bf16* wtb  = (__bf16*)carve((size_t)D4 * FF_DIM / 2 + 16); // transposed wt

    auto cvt = [&](const float* s, __bf16* dptr, size_t n) {
        int n8 = (int)(n / 8);
        f32_to_bf16_kernel<<<dim3((n8 + 255) / 256), dim3(256), 0, stream>>>(s, dptr, n8);
    };
    auto gemm = [&](const __bf16* Abf, const float* W, const float* bias,
                    float* C, int M, int K, int N, int relu) {
        transpose_bf16_kernel<<<dim3(K / 32, N / 32), dim3(256), 0, stream>>>(W, wtb, K, N);
        int mt = (M + 127) / 128;
        gemm_bf16_wmma<<<dim3(mt * (N / 32)), dim3(256), 0, stream>>>(Abf, wtb, bias, C, M, K, N, relu);
    };

    const int ND_BLK = (int)((ND + 255) / 256);
    const int EH_BLK = (int)((EH + 255) / 256);
    const int NH_BLK = (int)((NH + 255) / 256);

    // x_cur = x ; edge_attr -> bf16 once (constant across layers)
    hipMemcpyAsync(x_cur, x, ND * sizeof(float), hipMemcpyDeviceToDevice, stream);
    cvt(edge_attr, ebf, EDe);

    // node counts per graph
    hipMemsetAsync(cntb, 0, G_GR * sizeof(float), stream);
    count_kernel<<<dim3((N_NODES + 255) / 256), dim3(256), 0, stream>>>(batch, cntb);

    for (int l = 0; l < L_LAYERS; l++) {
        const float* Wq_l = Wq  + (size_t)l * D_DIM * D_DIM;
        const float* Wk_l = Wk  + (size_t)l * D_DIM * D_DIM;
        const float* Wv_l = Wv  + (size_t)l * D_DIM * D_DIM;
        const float* Ws_l = Wsk + (size_t)l * D_DIM * D_DIM;
        const float* We_l = We  + (size_t)l * DE_DIM * D_DIM;
        const float* bq_l = bq  + (size_t)l * D_DIM;
        const float* bk_l = bk  + (size_t)l * D_DIM;
        const float* bv_l = bv  + (size_t)l * D_DIM;
        const float* bs_l = bsk + (size_t)l * D_DIM;
        const float* be_l = be  + (size_t)l * D_DIM;

        cvt(x_cur, xbf, ND);   // shared bf16 A for the 4 node GEMMs

        gemm(xbf, Wq_l, bq_l, qb,    N_NODES, D_DIM, D_DIM, 0);
        gemm(xbf, Wk_l, bk_l, kb,    N_NODES, D_DIM, D_DIM, 0);
        gemm(xbf, Wv_l, bv_l, vb,    N_NODES, D_DIM, D_DIM, 0);
        gemm(xbf, Ws_l, bs_l, skipb, N_NODES, D_DIM, D_DIM, 0);
        gemm(ebf, We_l, be_l, eprojb, E_EDGES, DE_DIM, D_DIM, 0);

        fill_kernel<<<dim3(NH_BLK), dim3(256), 0, stream>>>(mb, -INFINITY, (int)NH);
        hipMemsetAsync(ssumb, 0, NH * sizeof(float), stream);
        hipMemsetAsync(aggb, 0, ND * sizeof(float), stream);

        edge_alpha_kernel<<<dim3(EH_BLK), dim3(256), 0, stream>>>(qb, kb, eprojb, src, dst, alphab, mb);
        edge_exp_kernel<<<dim3(EH_BLK), dim3(256), 0, stream>>>(alphab, mb, ssumb, dst);
        edge_agg_kernel<<<dim3(EH_BLK), dim3(256), 0, stream>>>(alphab, ssumb, vb, eprojb, src, dst, aggb);

        hipMemsetAsync(gsum, 0, GD * sizeof(float), stream);
        hipMemsetAsync(gvar, 0, GD * sizeof(float), stream);

        node_h_kernel<<<dim3(ND_BLK), dim3(256), 0, stream>>>(aggb, skipb, x_cur, hbuf, gsum, batch);
        gn_out_kernel<<<dim3(ND_BLK), dim3(256), 0, stream>>>(hbuf, gsum, cntb, gn_m + (size_t)l * D_DIM,
                                                              obuf, gvar, batch);
        gn_fin_kernel<<<dim3(ND_BLK), dim3(256), 0, stream>>>(obuf, gvar, cntb,
                                                              gn_w + (size_t)l * D_DIM,
                                                              gn_b + (size_t)l * D_DIM,
                                                              x_cur, batch);
    }

    // pooled mean
    hipMemsetAsync(pooled, 0, GD * sizeof(float), stream);
    pool_kernel<<<dim3(ND_BLK), dim3(256), 0, stream>>>(x_cur, pooled, batch);

    // RVQ (indices written straight into d_out region)
    hipMemsetAsync(lossb, 0, 8 * sizeof(float), stream);
    float* idxout = outp + GQF;
    rvq_kernel<<<dim3(G_GR), dim3(256), 0, stream>>>(pooled, cntb, codebooks, tokens, idxout, lossb);

    // token MLP (WMMA)
    cvt(tokens, tkbf, GQD);
    gemm(tkbf, W1,  b1,  h1b,   GQ, D_DIM, D4,     1);   // relu(tokens@W1+b1)
    cvt(h1b, h1bf, (size_t)GQ * D4);
    gemm(h1bf, W2,  b2,  projb, GQ, D4,    FF_DIM, 0);
    gemm(tkbf, Wsf, bsf, skfb,  GQ, D_DIM, FF_DIM, 0);

    add_ln_kernel<<<dim3(GQ),   dim3(256), 0, stream>>>(projb, skfb, ln_g, ln_b, tnorm);
    renorm_kernel<<<dim3(G_GR), dim3(256), 0, stream>>>(tnorm, kg_bias, outp);
    finalize_kernel<<<dim3(1), dim3(1), 0, stream>>>(lossb, outp + GQF + GQ);
}
